// CRF_687194768220
// MI455X (gfx1250) — matrix operand order repaired and checked by reading, not compiled, not run on hardware
//
#include <hip/hip_runtime.h>

typedef __attribute__((ext_vector_type(16))) _Float16 v16h;
typedef __attribute__((ext_vector_type(8)))  float    v8f;

#define B_ 512
#define S_ 512
#define T_ 96
#define LOG2E 1.4426950408889634f
#define LN2   0.6931471805599453f

// ---------------------------------------------------------------------------
// Kernel 1: gold-path score, one wave32 per batch element.
// ---------------------------------------------------------------------------
__global__ __launch_bounds__(256) void crf_score_kernel(
    const float* __restrict__ em, const int* __restrict__ tags,
    const unsigned char* __restrict__ mask, const float* __restrict__ sos,
    const float* __restrict__ trans, const float* __restrict__ eos,
    float* __restrict__ score_out)
{
    const int b    = (blockIdx.x * blockDim.x + threadIdx.x) >> 5;
    const int lane = threadIdx.x & 31;
    if (b >= B_) return;

    float part = 0.f;
    int   cnt  = 0;
    for (int s = lane; s < S_; s += 32) {
        int tg = tags[b * S_ + s];
        int mk = mask[b * S_ + s] ? 1 : 0;
        cnt += mk;
        if (s > 0 && !mk) {
            int tp = tags[b * S_ + s - 1];
            part += trans[tp * T_ + tg]
                  + em[(size_t)b * S_ * T_ + (size_t)s * T_ + tg];
        }
    }
    for (int off = 16; off > 0; off >>= 1) {
        part += __shfl_xor(part, off, 32);
        cnt  += __shfl_xor(cnt,  off, 32);
    }
    if (lane == 0) {
        int t0 = tags[b * S_];
        float sc = part + sos[t0] + em[(size_t)b * S_ * T_ + t0];
        int last = S_ - 1 - cnt;
        sc += eos[tags[b * S_ + last]];
        score_out[b] = sc;
    }
}

// ---------------------------------------------------------------------------
// Kernel 2: forward algorithm via exp-domain WMMA.
//   One workgroup = 16 batch rows, 6 waves; wave w owns output tags
//   [16w, 16w+16). B-fragments (exp(trans) in f16) live in VGPRs for the
//   whole 512-step scan; lp / P live in LDS. Row maxima via segmented LDS
//   partial maxes (no cross-lane permutes in the hot loop).
// ---------------------------------------------------------------------------
__global__ __launch_bounds__(192) void crf_forward_kernel(
    const float* __restrict__ em, const unsigned char* __restrict__ mask,
    const float* __restrict__ sos, const float* __restrict__ trans,
    const float* __restrict__ eos, const float* __restrict__ scores,
    float* __restrict__ out)
{
    __shared__ float    lp[16][T_];        // log-probs, fp32
    __shared__ _Float16 P[16][100];        // exp(lp - rowmax), f16 (padded stride)
    __shared__ float    tilemax[6][16];    // partial row maxima (6 x 16-col segs)
    __shared__ float    rowmax[16];        // per-row max of lp (for phase C)

    const int tid   = threadIdx.x;
    const int w     = tid >> 5;            // wave id: owns N-tile w
    const int lane  = tid & 31;
    const int half  = (lane >> 4) & 1;
    const int m16   = lane & 15;
    const int brow0 = blockIdx.x * 16;

    // Phase-B / partial-max thread mapping: row = tid&15, 12 col-segments of 8
    const int prow  = tid & 15;
    const int pcol0 = (tid >> 4) * 8;      // 0..88 step 8

    // ---- B fragments: E = exp(trans), constant across the scan -------------
    // B (32x16, KxN), contiguous k chosen here: lanes 0-15 hold K=0..15, lanes 16-31 K=16..31,
    // packed two K per VGPR; N = lane%16.
    v16h bf[3];
    #pragma unroll
    for (int kf = 0; kf < 3; ++kf) {
        #pragma unroll
        for (int v = 0; v < 8; ++v) {
            #pragma unroll
            for (int lo = 0; lo < 2; ++lo) {
                int K = kf * 32 + half * 16 + 2 * v + lo;
                int N = w * 16 + m16;
                float e = __builtin_amdgcn_exp2f(trans[K * T_ + N] * LOG2E);
                bf[kf][2 * v + lo] = (_Float16)e;
            }
        }
    }

    // ---- init: lp0 = sos + em[:, 0, :] -------------------------------------
    #pragma unroll
    for (int q = 0; q < 8; ++q) {
        int col = pcol0 + q;
        lp[prow][col] = sos[col] + em[(size_t)(brow0 + prow) * S_ * T_ + col];
    }
    __syncthreads();
    // initial segmented partial maxima
    if (tid < 96) {
        int row = tid & 15, seg = tid >> 4;
        float mx = lp[row][seg * 16];
        #pragma unroll
        for (int k = 1; k < 16; ++k) mx = fmaxf(mx, lp[row][seg * 16 + k]);
        tilemax[seg][row] = mx;
    }
    __syncthreads();

    // ---- scan over steps ----------------------------------------------------
    for (int s = 1; s < S_; ++s) {
        // Phase B: combine partial maxima privately, publish rowmax,
        //          P = exp(lp - rowmax) -> f16 in LDS
        {
            float rm = tilemax[0][prow];
            #pragma unroll
            for (int seg = 1; seg < 6; ++seg)
                rm = fmaxf(rm, tilemax[seg][prow]);
            if (tid < 16) rowmax[prow] = rm;     // for phase C (after barrier)
            #pragma unroll
            for (int q = 0; q < 8; ++q) {
                int col = pcol0 + q;
                P[prow][col] =
                    (_Float16)__builtin_amdgcn_exp2f((lp[prow][col] - rm) * LOG2E);
            }
        }
        __syncthreads();

        // Phase C: Q = P x E via 3 chained v_wmma_f32_16x16x32_f16
        v8f acc = {};
        #pragma unroll
        for (int kf = 0; kf < 3; ++kf) {
            // 16-bit A (16x32, MxK): M = lane%16; K = 16*half + 2v (+lo).
            // Contiguous per half ->
            // compiler merges into ds_load_b128 pairs.
            v16h a;
            #pragma unroll
            for (int v = 0; v < 8; ++v) {
                int K = kf * 32 +
                        (half * 16 + 2 * v);
                a[2 * v]     = P[m16][K];
                a[2 * v + 1] = P[m16][K + 1];
            }
            acc = __builtin_amdgcn_wmma_f32_16x16x32_f16(
                false, a, false, bf[kf], (short)0, acc, false, false);
        }

        // lp_new = rowmax + log(Q) + em[:, s, :]  (masked rows keep old lp)
        #pragma unroll
        for (int r = 0; r < 8; ++r) {
            int M    = r + half * 8;       // C/D row layout
            int brow = brow0 + M;
            int j    = w * 16 + m16;
            float oldv = lp[M][j];
            unsigned char mk = mask[brow * S_ + s];
            float val = rowmax[M]
                      + __builtin_amdgcn_logf(acc[r]) * LN2
                      + em[(size_t)brow * S_ * T_ + (size_t)s * T_ + j];
            lp[M][j] = mk ? oldv : val;
            if (s + 1 < S_)                // stream next step's emissions
                __builtin_prefetch(&em[(size_t)brow * S_ * T_ +
                                       (size_t)(s + 1) * T_ + j], 0, 1);
        }
        __syncthreads();

        // Phase A: segmented partial row maxima of new lp (pipelined ds loads,
        // no cross-lane permutes)
        if (tid < 96) {
            int row = tid & 15, seg = tid >> 4;
            float mx = lp[row][seg * 16];
            #pragma unroll
            for (int k = 1; k < 16; ++k) mx = fmaxf(mx, lp[row][seg * 16 + k]);
            tilemax[seg][row] = mx;
        }
        __syncthreads();
    }

    // ---- finish: logZ = logsumexp(lp + eos); out = logZ - gold score -------
    if (tid < 16) {
        int row = tid;
        float mx = -3.4e38f;
        for (int j = 0; j < T_; ++j) mx = fmaxf(mx, lp[row][j] + eos[j]);
        float sum = 0.f;
        for (int j = 0; j < T_; ++j)
            sum += __builtin_amdgcn_exp2f((lp[row][j] + eos[j] - mx) * LOG2E);
        float logZ = mx + __builtin_amdgcn_logf(sum) * LN2;
        out[brow0 + row] = logZ - scores[brow0 + row];
    }
}

// ---------------------------------------------------------------------------
extern "C" void kernel_launch(void* const* d_in, const int* in_sizes, int n_in,
                              void* d_out, int out_size, void* d_ws, size_t ws_size,
                              hipStream_t stream)
{
    const float*         em    = (const float*)d_in[0];
    const int*           tags  = (const int*)d_in[1];
    const unsigned char* mask  = (const unsigned char*)d_in[2];
    const float*         sos   = (const float*)d_in[3];
    const float*         trans = (const float*)d_in[4];
    const float*         eos   = (const float*)d_in[5];
    float* out    = (float*)d_out;
    float* scores = (float*)d_ws;   // B_ floats of scratch

    crf_score_kernel<<<B_ / 8, 256, 0, stream>>>(em, tags, mask, sos, trans,
                                                 eos, scores);
    crf_forward_kernel<<<B_ / 16, 192, 0, stream>>>(em, mask, sos, trans, eos,
                                                    scores, out);
}
